// PatchCoreInteraction_5463198400802
// MI455X (gfx1250) — matrix operand order repaired and checked by reading, not compiled, not run on hardware
//
#include <hip/hip_runtime.h>
#include <hip/hip_bf16.h>
#include <cstdint>
#include <cstddef>

// ---------------------------------------------------------------------------
// MI455X (gfx1250). All GEMMs on v_wmma_f32_16x16x32_bf16 (wave32, f32 acc).
// Staging uses GLOBAL_LOAD_ASYNC_TO_LDS_B128 when the builtin is available.
// ---------------------------------------------------------------------------

typedef __attribute__((ext_vector_type(16))) __bf16    bf16x16;
typedef __attribute__((ext_vector_type(8)))  float     f32x8;
typedef __attribute__((ext_vector_type(4)))  float     f32x4;
typedef __attribute__((ext_vector_type(4)))  unsigned  u32x4;
typedef __attribute__((ext_vector_type(4)))  int       i32x4;

using u16 = unsigned short;
using u32 = unsigned int;
using s64 = long long;

#define NB   16
#define NP   2048
#define ND   512
#define NH   8
#define NC   64
#define NHD  64
#define FIN  1024
#define FHID 4096

#if defined(__has_builtin)
#if __has_builtin(__builtin_amdgcn_global_load_async_to_lds_b128)
#define HAVE_ASYNC_LDS 1
#endif
#endif

__device__ __forceinline__ u16 f2bf(float f) {
    u32 u = __builtin_bit_cast(u32, f);
    u32 r = u + 0x7FFFu + ((u >> 16) & 1u);          // round-to-nearest-even
    if ((u & 0x7F800000u) == 0x7F800000u) r = u;     // inf/nan: truncate
    return (u16)(r >> 16);
}

// 16-byte global->LDS copy: async on CDNA5 when available.
__device__ __forceinline__ void cp_b128(const u16* g, u16* l) {
#ifdef HAVE_ASYNC_LDS
    __builtin_amdgcn_global_load_async_to_lds_b128((i32x4*)g, (i32x4*)l, 0, 0);
#else
    *(u32x4*)l = *(const u32x4*)g;
#endif
}
__device__ __forceinline__ void wait_async_stage() {
#ifdef HAVE_ASYNC_LDS
#if __has_builtin(__builtin_amdgcn_s_wait_asynccnt)
    __builtin_amdgcn_s_wait_asynccnt(0);
#else
    asm volatile("s_wait_asynccnt 0x0" ::: "memory");
#endif
#endif
}

// ------------------------------ convert ------------------------------------
__global__ void cvt_f32_bf16(const float* __restrict__ src, u16* __restrict__ dst, s64 n) {
    s64 i = (s64)blockIdx.x * blockDim.x + threadIdx.x;
    if (i < n) dst[i] = f2bf(src[i]);
}

// ------------------------------ GEMM ---------------------------------------
// C[M,N] = A[M,K]*B[K,N] (bf16 in, f32 acc), batch z: zo=z/mod, zi=z%mod.
// BL: 0 = B row-major (k*ldb + n), 1 = B K-contiguous (n*ldb + k).
// EPI: 0 f32*scale | 1 bf16(+bias) | 2 bf16 bias+GELU | 3 vpatch dual write
//      4 f32 bias+resid
// GUARD: false when M%BM==0 and N%BN==0 (no bounds checks anywhere).
struct GemmP {
    const u16* A;  s64 a_outer, a_inner; int lda;
    const u16* Bm; s64 b_outer, b_inner; int ldb;
    void*      C;  s64 c_outer, c_inner; int ldc;
    const float* bias;
    const float* resid; s64 r_outer, r_inner; int ldr;
    int M, N, K, mod;
    float scale;
    int out2_off;
};

#define BM 128
#define BN 128
#define BK 32
#define LDS_STRIDE 40   // 32 + 8 bf16 pad; keeps 16B alignment

template <int BL, int EPI, bool GUARD>
__global__ __launch_bounds__(256) void gemm_wmma(GemmP p) {
    __shared__ u16 lA[BM * LDS_STRIDE];
    __shared__ u16 lB[BN * LDS_STRIDE];

    const int z  = blockIdx.z;
    const int zo = z / p.mod, zi = z % p.mod;
    const u16* A  = p.A  + zo * p.a_outer + zi * p.a_inner;
    const u16* Bg = p.Bm + zo * p.b_outer + zi * p.b_inner;

    const int m0 = blockIdx.y * BM;
    const int n0 = blockIdx.x * BN;
    const int tid  = threadIdx.x;
    const int lane = tid & 31;
    const int wave = tid >> 5;
    const int wmBase = (wave & 3) * 32;   // 4 waves along M
    const int wnBase = (wave >> 2) * 64;  // 2 waves along N

    f32x8 acc[2][4];
#pragma unroll
    for (int mi = 0; mi < 2; ++mi)
#pragma unroll
        for (int ni = 0; ni < 4; ++ni)
            acc[mi][ni] = (f32x8){0.f,0.f,0.f,0.f,0.f,0.f,0.f,0.f};

    union FragCvt { u32x4 q[2]; bf16x16 v; };
    union SegCvt  { u32x4 q[2]; u16 h[16]; };

    // per-thread staging coordinates
    const int arow = tid >> 1, ahalf = tid & 1;           // A: 128 rows x 2 chunks

    for (int k0 = 0; k0 < p.K; k0 += BK) {
        __syncthreads();
        // ---- stage A tile (BM x BK), row-major, 32B per thread
        {
            const int gm = m0 + arow;
            u16* d = &lA[arow * LDS_STRIDE + ahalf * 16];
            if (!GUARD || gm < p.M) {
                const u16* s = A + (s64)gm * p.lda + k0 + ahalf * 16;
                cp_b128(s, d);
                cp_b128(s + 8, d + 8);
                if (k0 + BK < p.K) __builtin_prefetch(s + BK, 0, 1);
            } else {
                const u32x4 zz = (u32x4){0,0,0,0};
                *(u32x4*)d = zz; *(u32x4*)(d + 8) = zz;
            }
        }
        // ---- stage B tile, transposed into LDS as [n][k]
        if (BL == 0) {
            // B row-major: coalesced b128 reads, b16 transpose scatter to LDS.
            const int kk = tid >> 3, nseg = tid & 7;
            const int gn0 = n0 + nseg * 16;
            SegCvt sc;
            sc.q[0] = (u32x4){0,0,0,0}; sc.q[1] = sc.q[0];
            if (!GUARD || gn0 + 16 <= p.N) {               // N is a multiple of 16
                const u16* s = Bg + (s64)(k0 + kk) * p.ldb + gn0;
                sc.q[0] = *(const u32x4*)s;
                sc.q[1] = *(const u32x4*)(s + 8);
                if (k0 + BK < p.K) __builtin_prefetch(s + (s64)BK * p.ldb, 0, 1);
            }
#pragma unroll
            for (int i = 0; i < 16; ++i)
                lB[(nseg * 16 + i) * LDS_STRIDE + kk] = sc.h[i];
        } else {
            // B K-contiguous: direct (async) copy, already [n][k].
            const int col = tid >> 1, half = tid & 1;
            const int gn  = n0 + col;
            u16* d = &lB[col * LDS_STRIDE + half * 16];
            if (!GUARD || gn < p.N) {
                const u16* s = Bg + (s64)gn * p.ldb + k0 + half * 16;
                cp_b128(s, d);
                cp_b128(s + 8, d + 8);
            } else {
                const u32x4 zz = (u32x4){0,0,0,0};
                *(u32x4*)d = zz; *(u32x4*)(d + 8) = zz;
            }
        }
        wait_async_stage();
        __syncthreads();

        // ---- fragments (ISA 7.12.2 16-bit layouts) and WMMA
        const int lm     = lane & 15;
        const int khalfA = (lane >> 4) * 8;    // A: lanes16-31 hold K+8
        const int kbaseB = (lane >> 4) * 8;    // B: lanes16-31 hold K+8

        bf16x16 aF[2], bF[4];
#pragma unroll
        for (int mi = 0; mi < 2; ++mi) {
            const int row = wmBase + mi * 16 + lm;
            FragCvt u;
            u.q[0] = *(const u32x4*)&lA[row * LDS_STRIDE + khalfA];
            u.q[1] = *(const u32x4*)&lA[row * LDS_STRIDE + 16 + khalfA];
            aF[mi] = u.v;
        }
#pragma unroll
        for (int ni = 0; ni < 4; ++ni) {
            const int col = wnBase + ni * 16 + lm;
            FragCvt u;
            u.q[0] = *(const u32x4*)&lB[col * LDS_STRIDE + kbaseB];
            u.q[1] = *(const u32x4*)&lB[col * LDS_STRIDE + 16 + kbaseB];
            bF[ni] = u.v;
        }
#pragma unroll
        for (int mi = 0; mi < 2; ++mi)
#pragma unroll
            for (int ni = 0; ni < 4; ++ni)
                acc[mi][ni] = __builtin_amdgcn_wmma_f32_16x16x32_bf16(
                    false, aF[mi], false, bF[ni], (short)0, acc[mi][ni], false, false);
    }

    // ---- epilogue. C/D layout: N = lane&15, M = vgpr + 8*(lane>=16)
    const s64 co = (s64)zo * p.c_outer + (s64)zi * p.c_inner;
    const s64 ro = (s64)zo * p.r_outer + (s64)zi * p.r_inner;
    const int lm = lane & 15;
    const int mh = (lane >> 4) * 8;
#pragma unroll
    for (int ni = 0; ni < 4; ++ni) {
        const int n = n0 + wnBase + ni * 16 + lm;
        const bool nok = !GUARD || (n < p.N);
        float bn = 0.f;
        if (EPI == 1 || EPI == 2 || EPI == 4)
            if (nok && p.bias) bn = p.bias[n];
#pragma unroll
        for (int mi = 0; mi < 2; ++mi)
#pragma unroll
            for (int j = 0; j < 8; ++j) {
                const int m = m0 + wmBase + mi * 16 + mh + j;
                if (GUARD && (!nok || m >= p.M)) continue;
                float v = acc[mi][ni][j];
                const s64 idx = co + (s64)m * p.ldc + n;
                if (EPI == 0) {
                    ((float*)p.C)[idx] = v * p.scale;
                } else if (EPI == 1) {
                    ((u16*)p.C)[idx] = f2bf(v + bn);
                } else if (EPI == 2) {
                    v += bn;
                    v = 0.5f * v * (1.0f + erff(v * 0.70710678118654752f)); // exact GELU
                    ((u16*)p.C)[idx] = f2bf(v);
                } else if (EPI == 3) {
                    const float inp = p.resid[ro + (s64)m * p.ldr + n];
                    u16* o = (u16*)p.C;
                    o[idx]              = f2bf(inp - v);   // input - v_out
                    o[idx + p.out2_off] = f2bf(v);         // v_out
                } else { // EPI == 4
                    v += bn + p.resid[ro + (s64)m * p.ldr + n];
                    ((float*)p.C)[idx] = v;
                }
            }
    }
}

// --------------------- softmax over patches (axis = p) ---------------------
__global__ __launch_bounds__(256) void softmax_rows(const float* __restrict__ aff,
                                                    u16* __restrict__ out, int rowlen) {
    const s64 row = blockIdx.x;
    const float* src = aff + row * (s64)rowlen;
    __shared__ float red[256];
    const int t = threadIdx.x;

    float mx = -3.4e38f;
    for (int i = t; i < rowlen; i += 256) mx = fmaxf(mx, src[i]);
    red[t] = mx; __syncthreads();
    for (int s = 128; s > 0; s >>= 1) { if (t < s) red[t] = fmaxf(red[t], red[t + s]); __syncthreads(); }
    mx = red[0]; __syncthreads();

    float sum = 0.f;
    for (int i = t; i < rowlen; i += 256) sum += expf(src[i] - mx);
    red[t] = sum; __syncthreads();
    for (int s = 128; s > 0; s >>= 1) { if (t < s) red[t] += red[t + s]; __syncthreads(); }
    const float inv = 1.0f / red[0];

    u16* o = out + row * (s64)rowlen;
    for (int i = t; i < rowlen; i += 256) o[i] = f2bf(expf(src[i] - mx) * inv);
}

// ----------- softmax over cores (axis = c), output transposed (p,c) --------
__global__ void softmax_cols(const float* __restrict__ aff, u16* __restrict__ outT) {
    const s64 idx = (s64)blockIdx.x * blockDim.x + threadIdx.x;   // over BH*P
    if (idx >= (s64)NB * NH * NP) return;
    const s64 bh = idx / NP;
    const int pp = (int)(idx % NP);
    const float* src = aff + bh * (s64)NC * NP + pp;              // stride P over c

    float mx = -3.4e38f;
    for (int c = 0; c < NC; ++c) mx = fmaxf(mx, src[(s64)c * NP]);
    float sum = 0.f;
    for (int c = 0; c < NC; ++c) sum += expf(src[(s64)c * NP] - mx);
    const float inv = 1.0f / sum;

    u16* o = outT + (bh * NP + pp) * NC;
    for (int c = 0; c < NC; ++c) o[c] = f2bf(expf(src[(s64)c * NP] - mx) * inv);
}

// ------------------------------ LayerNorm ----------------------------------
__global__ __launch_bounds__(128) void layernorm_rows(const float* __restrict__ x,
                                                      const float* __restrict__ gamma,
                                                      const float* __restrict__ beta,
                                                      float* __restrict__ out) {
    const s64 row = blockIdx.x;
    const int t = threadIdx.x;                       // 128 threads * 4 floats = 512
    const f32x4 v = ((const f32x4*)(x + row * ND))[t];
    __shared__ float red[128];

    red[t] = v.x + v.y + v.z + v.w; __syncthreads();
    for (int s = 64; s > 0; s >>= 1) { if (t < s) red[t] += red[t + s]; __syncthreads(); }
    const float mean = red[0] * (1.0f / ND); __syncthreads();

    const float d0 = v.x - mean, d1 = v.y - mean, d2 = v.z - mean, d3 = v.w - mean;
    red[t] = d0 * d0 + d1 * d1 + d2 * d2 + d3 * d3; __syncthreads();
    for (int s = 64; s > 0; s >>= 1) { if (t < s) red[t] += red[t + s]; __syncthreads(); }
    const float inv = rsqrtf(red[0] * (1.0f / ND) + 1e-5f);

    const f32x4 g = ((const f32x4*)gamma)[t];
    const f32x4 b = ((const f32x4*)beta)[t];
    f32x4 o;
    o.x = d0 * inv * g.x + b.x;
    o.y = d1 * inv * g.y + b.y;
    o.z = d2 * inv * g.z + b.z;
    o.w = d3 * inv * g.w + b.w;
    ((f32x4*)(out + row * ND))[t] = o;
}

// ------------------------------ launch -------------------------------------
extern "C" void kernel_launch(void* const* d_in, const int* in_sizes, int n_in,
                              void* d_out, int out_size, void* d_ws, size_t ws_size,
                              hipStream_t stream) {
    const float* input = (const float*)d_in[0];
    const float* cores = (const float*)d_in[1];
    const float* Wv    = (const float*)d_in[2];
    const float* bv    = (const float*)d_in[3];
    const float* W1    = (const float*)d_in[4];
    const float* b1    = (const float*)d_in[5];
    const float* W2    = (const float*)d_in[6];
    const float* b2    = (const float*)d_in[7];
    const float* gamma = (const float*)d_in[8];
    const float* beta  = (const float*)d_in[9];
    float* outp = (float*)d_out;

    size_t off = 0;
    auto alloc = [&](size_t bytes) -> char* {
        size_t o = (off + 255) & ~(size_t)255;
        off = o + bytes;
        return (char*)d_ws + o;
    };
    const s64 nIn   = (s64)NB * NP * ND;        // 16.78M
    const s64 nAff  = (s64)NB * NH * NC * NP;   // 16.78M
    const s64 nFfn  = (s64)NB * NP * FIN;       // 33.55M
    const s64 nHid  = (s64)NB * NP * FHID;      // 134.2M

    u16*  in_bf    = (u16*)alloc(nIn * 2);
    u16*  q_bf     = (u16*)alloc(nIn * 2);
    u16*  cores_bf = (u16*)alloc((s64)NH * NC * NHD * 2);
    u16*  Wv_bf    = (u16*)alloc((s64)ND * ND * 2);
    u16*  W1_bf    = (u16*)alloc((s64)FIN * FHID * 2);
    u16*  W2_bf    = (u16*)alloc((s64)FHID * ND * 2);
    float* aff     = (float*)alloc(nAff * 4);   // reused as out_pre after softmaxes
    u16*  A1_bf    = (u16*)alloc(nAff * 2);
    u16*  A2t_bf   = (u16*)alloc(nAff * 2);
    u16*  vcore_bf = (u16*)alloc((s64)NB * NH * NC * NHD * 2);
    u16*  ffn_bf   = (u16*)alloc(nFfn * 2);
    u16*  hid_bf   = (u16*)alloc(nHid * 2);
    float* out_pre = aff;

    // ---- conversions to bf16
    auto cvt = [&](const float* s, u16* d, s64 n) {
        cvt_f32_bf16<<<dim3((unsigned)((n + 255) / 256)), dim3(256), 0, stream>>>(s, d, n);
    };
    cvt(input, in_bf, nIn);
    cvt(cores, cores_bf, (s64)NH * NC * NHD);
    cvt(Wv, Wv_bf, (s64)ND * ND);
    cvt(W1, W1_bf, (s64)FIN * FHID);
    cvt(W2, W2_bf, (s64)FHID * ND);

    GemmP p;
    auto clr = [&]() { p = GemmP{}; p.mod = 1; p.scale = 1.0f; };

    // ---- q = in*Wv + bv   (M=32768, N=512, K=512) -> bf16 (b,p,h,d)
    clr();
    p.A = in_bf; p.lda = ND;
    p.Bm = Wv_bf; p.ldb = ND;
    p.C = q_bf; p.ldc = ND;
    p.bias = bv;
    p.M = NB * NP; p.N = ND; p.K = ND;
    gemm_wmma<0, 1, false><<<dim3(ND / BN, (NB * NP) / BM, 1), dim3(256), 0, stream>>>(p);

    // ---- aff[b,h] = cores[h] * q[b,:,h,:]^T / 8   (M=64, N=2048, K=64) x128
    clr();
    p.A = cores_bf; p.a_outer = 0; p.a_inner = (s64)NC * NHD; p.lda = NHD;
    p.Bm = q_bf; p.b_outer = (s64)NP * ND; p.b_inner = NHD; p.ldb = ND;   // K-contig
    p.C = aff; p.c_outer = (s64)NH * NC * NP; p.c_inner = (s64)NC * NP; p.ldc = NP;
    p.M = NC; p.N = NP; p.K = NHD; p.mod = NH; p.scale = 0.125f;
    gemm_wmma<1, 0, true><<<dim3(NP / BN, 1, NB * NH), dim3(256), 0, stream>>>(p);

    // ---- dual softmax
    softmax_rows<<<dim3(NB * NH * NC), dim3(256), 0, stream>>>(aff, A1_bf, NP);
    softmax_cols<<<dim3((unsigned)(((s64)NB * NH * NP + 255) / 256)), dim3(256), 0, stream>>>(aff, A2t_bf);

    // ---- v_core[b,h] = A1[b,h] * v[b,h]   (M=64, N=64, K=2048) x128
    clr();
    p.A = A1_bf; p.a_outer = (s64)NH * NC * NP; p.a_inner = (s64)NC * NP; p.lda = NP;
    p.Bm = in_bf; p.b_outer = (s64)NP * ND; p.b_inner = NHD; p.ldb = ND;
    p.C = vcore_bf; p.c_outer = (s64)NH * NC * NHD; p.c_inner = (s64)NC * NHD; p.ldc = NHD;
    p.M = NC; p.N = NHD; p.K = NP; p.mod = NH;
    gemm_wmma<0, 1, true><<<dim3(1, 1, NB * NH), dim3(256), 0, stream>>>(p);

    // ---- v_patch[b,h] = A2^T[b,h] * v_core[b,h]  (M=2048, N=64, K=64) x128
    //      fused: ffn_in[:, :512]=in - v_out ; ffn_in[:, 512:]=v_out  (bf16)
    clr();
    p.A = A2t_bf; p.a_outer = (s64)NH * NP * NC; p.a_inner = (s64)NP * NC; p.lda = NC;
    p.Bm = vcore_bf; p.b_outer = (s64)NH * NC * NHD; p.b_inner = (s64)NC * NHD; p.ldb = NHD;
    p.C = ffn_bf; p.c_outer = (s64)NP * FIN; p.c_inner = NHD; p.ldc = FIN; p.out2_off = ND;
    p.resid = input; p.r_outer = (s64)NP * ND; p.r_inner = NHD; p.ldr = ND;
    p.M = NP; p.N = NHD; p.K = NC; p.mod = NH;
    gemm_wmma<0, 3, true><<<dim3(1, NP / BM, NB * NH), dim3(256), 0, stream>>>(p);

    // ---- hid = gelu(ffn_in*W1 + b1)   (M=32768, N=4096, K=1024)
    clr();
    p.A = ffn_bf; p.lda = FIN;
    p.Bm = W1_bf; p.ldb = FHID;
    p.C = hid_bf; p.ldc = FHID;
    p.bias = b1;
    p.M = NB * NP; p.N = FHID; p.K = FIN;
    gemm_wmma<0, 2, false><<<dim3(FHID / BN, (NB * NP) / BM, 1), dim3(256), 0, stream>>>(p);

    // ---- out_pre = input + hid*W2 + b2   (M=32768, N=512, K=4096)
    clr();
    p.A = hid_bf; p.lda = FHID;
    p.Bm = W2_bf; p.ldb = ND;
    p.C = out_pre; p.ldc = ND;
    p.bias = b2;
    p.resid = input; p.ldr = ND;
    p.M = NB * NP; p.N = ND; p.K = FHID;
    gemm_wmma<0, 4, false><<<dim3(ND / BN, (NB * NP) / BM, 1), dim3(256), 0, stream>>>(p);

    // ---- LayerNorm -> d_out
    layernorm_rows<<<dim3(NB * NP), dim3(128), 0, stream>>>(out_pre, gamma, beta, outp);
}
